// VisualSudokuModule_5677946766050
// MI455X (gfx1250) — matrix operand order repaired and checked by reading, not compiled, not run on hardware
//
#include <hip/hip_runtime.h>
#include <hip/hip_bf16.h>
#include <cstddef>

// ---------------------------------------------------------------------------
// Visual-Sudoku pipeline for MI455X (gfx1250, wave32, WMMA).
//
// Stage 1: LeNet on 20736 digit images.
//   conv1+pool+relu (VALU, LDS-staged)            -> act1  f32 [N][864]
//   conv2+pool+relu (VALU, LDS-staged)            -> act2h f16 [N][256]
//   fc1 via v_wmma_f32_16x16x32_f16               -> fc1oh f16 [N][128] (pad 0)
//   fc2 via v_wmma_f32_16x16x32_f16               -> fc2o  f32 [N][84]
//   fc3 + log_softmax (VALU)                      -> lp    f32 [N][9]
//   Weights pre-converted once to zero-padded f16 so all WMMA fragment loads
//   are unconditional global_load_b128 (no per-element exec-mask branches).
// Stage 2: log-domain circuit, values BATCH-MAJOR [W][256]: one block per
//   node, threadIdx.x = batch column -> block-uniform scalar index loads and
//   fully coalesced 1KB child-row gathers (L2-resident: 32MB layer << 192MB).
//
// Workspace layout (bytes):
//   CBUF0 @ 0           33,554,432  } both alias ACT1 (f32 [N][864],
//   CBUF1 @ 33,554,432  33,554,432  }  0..71,663,616, dead after conv2)
//   ACT2H @ 71,663,616  10,616,832  (f16 [N][256])
//   FC1OH @ 82,280,448   5,308,416  (f16 [N][128])
//   FC2O  @ 87,588,864   6,967,296  (f32 [N][84])
//   WH1   @ 94,556,160      65,536  (f16 [128][256])
//   WH2   @ 94,621,696      24,576  (f16 [96][128])
//   LP    @ 94,646,272     746,496  (f32 [N][9])
//   total required ws_size >= 95,392,768 bytes (~91 MiB)
// ---------------------------------------------------------------------------

typedef __attribute__((ext_vector_type(16))) _Float16 v16h;
typedef __attribute__((ext_vector_type(8)))  _Float16 v8h;
typedef __attribute__((ext_vector_type(8)))  float    v8f;

#define NIMG 20736

// ------------------------- conv1 + pool + relu -----------------------------
__global__ void conv1_pool_kernel(const float* __restrict__ images,
                                  const float* __restrict__ w,
                                  const float* __restrict__ b,
                                  float* __restrict__ act1)
{
    __shared__ float simg[784];
    __shared__ float sw[150];
    __shared__ float sb[6];
    const int img = blockIdx.x;
    const float* ip = images + (size_t)img * 784;
    for (int j = threadIdx.x; j < 784; j += blockDim.x) simg[j] = ip[j];
    for (int j = threadIdx.x; j < 150; j += blockDim.x) sw[j]  = w[j];
    if (threadIdx.x < 6) sb[threadIdx.x] = b[threadIdx.x];
    __syncthreads();

    for (int o = threadIdx.x; o < 6 * 12 * 12; o += blockDim.x) {
        const int c  = o / 144;
        const int r  = o % 144;
        const int oy = r / 12, ox = r % 12;
        const float* wc = sw + c * 25;
        float mx = -3.4e38f;
        #pragma unroll
        for (int dy = 0; dy < 2; ++dy)
        #pragma unroll
        for (int dx = 0; dx < 2; ++dx) {
            const int iy = oy * 2 + dy, ix = ox * 2 + dx;
            float s = sb[c];
            #pragma unroll
            for (int ky = 0; ky < 5; ++ky)
            #pragma unroll
            for (int kx = 0; kx < 5; ++kx)
                s += simg[(iy + ky) * 28 + ix + kx] * wc[ky * 5 + kx];
            mx = fmaxf(mx, s);
        }
        act1[(size_t)img * 864 + o] = fmaxf(mx, 0.f);
    }
}

// ----------------- conv2 + pool + relu, f16 output -------------------------
__global__ void conv2_pool_kernel(const float* __restrict__ act1,
                                  const float* __restrict__ w,
                                  const float* __restrict__ b,
                                  _Float16* __restrict__ act2h)
{
    __shared__ float sin_[864];    // 6 x 12 x 12
    __shared__ float sw[2400];     // 16 x 6 x 5 x 5
    __shared__ float sb[16];
    const int img = blockIdx.x;
    const float* ip = act1 + (size_t)img * 864;
    for (int j = threadIdx.x; j < 864;  j += blockDim.x) sin_[j] = ip[j];
    for (int j = threadIdx.x; j < 2400; j += blockDim.x) sw[j]   = w[j];
    if (threadIdx.x < 16) sb[threadIdx.x] = b[threadIdx.x];
    __syncthreads();

    // 16*4*4 = 256 outputs: exactly one per thread
    const int o  = threadIdx.x;
    const int c  = o / 16;
    const int r  = o % 16;
    const int oy = r / 4, ox = r % 4;
    float mx = -3.4e38f;
    #pragma unroll
    for (int dy = 0; dy < 2; ++dy)
    #pragma unroll
    for (int dx = 0; dx < 2; ++dx) {
        const int iy = oy * 2 + dy, ix = ox * 2 + dx;
        float s = sb[c];
        for (int ic = 0; ic < 6; ++ic) {
            const float* wc = sw + ((c * 6 + ic) * 25);
            const float* pp = sin_ + ic * 144;
            #pragma unroll
            for (int ky = 0; ky < 5; ++ky)
            #pragma unroll
            for (int kx = 0; kx < 5; ++kx)
                s += pp[(iy + ky) * 12 + ix + kx] * wc[ky * 5 + kx];
        }
        mx = fmaxf(mx, s);
    }
    act2h[(size_t)img * 256 + o] = (_Float16)fmaxf(mx, 0.f);
}

// ------- one-shot weight conversion: f32 -> zero-padded f16 matrices -------
// wh1: [128][256] from fc1_w [120][256] (rows 120..127 = 0)
// wh2: [96][128]  from fc2_w [84][120]  (rows 84..95 = 0, cols 120..127 = 0)
__global__ void weights_to_f16_kernel(const float* __restrict__ w1,
                                      const float* __restrict__ w2,
                                      _Float16* __restrict__ wh1,
                                      _Float16* __restrict__ wh2)
{
    const int tid = blockIdx.x * blockDim.x + threadIdx.x;
    const int nt  = gridDim.x * blockDim.x;
    for (int i = tid; i < 128 * 256; i += nt) {
        const int n = i >> 8, k = i & 255;
        wh1[i] = (n < 120) ? (_Float16)w1[n * 256 + k] : (_Float16)0.f;
    }
    for (int i = tid; i < 96 * 128; i += nt) {
        const int n = i >> 7, k = i & 127;
        wh2[i] = (n < 84 && k < 120) ? (_Float16)w2[n * 120 + k] : (_Float16)0.f;
    }
}

// ------------------------- FC via WMMA f16->f32 ----------------------------
// A : f16 [M][KPAD] row-major, zero-padded. Wh: f16 [>=NTILES*16][KPAD],
// zero-padded. One wave computes one 16x16 tile; all fragment loads are
// unconditional vector loads. blockDim.x == 32 * NTILES exactly.
// Fragment layouts used here (wave32):
//   A 16x32 f16 : lane<16 -> M=lane, K = k0..k0+15 (contiguous);
//                 lane>=16 -> M=lane-16, K = k0+16..31.
//   B 32x16 f16 : N = lane&15; this code's K order, same as A: lane<16 K=k0..k0+15, lane>=16 K=k0+16..31.
//   C/D 16x16 f32: N = lane&15, M = vgpr + 8*(lane>>4).
template<int KPAD, int NOUT, int OUT_STRIDE, bool RELU, typename OutT>
__global__ void fc_wmma_kernel(const _Float16* __restrict__ A,
                               const _Float16* __restrict__ Wh,
                               const float* __restrict__ bias,
                               OutT* __restrict__ out)
{
    constexpr int KTILES = KPAD / 32;
    const int wave = threadIdx.x >> 5;
    const int lane = threadIdx.x & 31;
    const int m0   = blockIdx.x * 16;
    const int n0   = wave * 16;
    const int half = lane >> 4;
    const int l15  = lane & 15;

    const _Float16* __restrict__ pa = A  + (size_t)(m0 + l15) * KPAD;
    const _Float16* __restrict__ pw = Wh + (size_t)(n0 + l15) * KPAD;

    v8f acc = {0.f, 0.f, 0.f, 0.f, 0.f, 0.f, 0.f, 0.f};

    #pragma unroll
    for (int kt = 0; kt < KTILES; ++kt) {
        const int k0 = kt * 32;
        const v16h af = *(const v16h*)(pa + k0 + half * 16);      // 32B load
        const v16h bf = *(const v16h*)(pw + k0 + half * 16);      // 32B load
        acc = __builtin_amdgcn_wmma_f32_16x16x32_f16(
                  false, af, false, bf, (short)0, acc, false, false);
    }

    const int n = n0 + l15;
    const float bs = (n < NOUT) ? bias[n] : 0.f;
    #pragma unroll
    for (int r = 0; r < 8; ++r) {
        const int m = m0 + half * 8 + r;
        float v = acc[r] + bs;
        if (RELU) v = fmaxf(v, 0.f);
        if (OUT_STRIDE > NOUT) {                 // padded output: store all lanes
            out[(size_t)m * OUT_STRIDE + n] = (OutT)v;
        } else if (n < NOUT) {                   // exact output: mask pad lanes
            out[(size_t)m * OUT_STRIDE + n] = (OutT)v;
        }
    }
}

// ------------------------- FC3 + log_softmax -------------------------------
__global__ void fc3_lsm_kernel(const float* __restrict__ A,
                               const float* __restrict__ W,
                               const float* __restrict__ b,
                               float* __restrict__ lp)
{
    const int i = blockIdx.x * blockDim.x + threadIdx.x;
    if (i >= NIMG) return;
    const float* a = A + (size_t)i * 84;
    float z[9];
    float m = -3.4e38f;
    #pragma unroll
    for (int n = 0; n < 9; ++n) {
        float s = b[n];
        const float* wn = W + n * 84;
        for (int k = 0; k < 84; ++k) s += a[k] * wn[k];
        z[n] = s;
        m = fmaxf(m, s);
    }
    float e = 0.f;
    #pragma unroll
    for (int n = 0; n < 9; ++n) e += __expf(z[n] - m);
    const float lse = m + __logf(e);
    #pragma unroll
    for (int n = 0; n < 9; ++n) lp[(size_t)i * 9 + n] = z[n] - lse;
}

// ---------------- transpose leaf values to batch-major [1458][256] ---------
__global__ void leaf_kernel(const float* __restrict__ lp, float* __restrict__ leaf)
{
    const int l  = blockIdx.x;   // 0..1457
    const int bb = threadIdx.x;  // 0..255 (batch column)
    leaf[(size_t)l * 256 + bb] = (l < 729) ? lp[(size_t)bb * 729 + l] : 0.f;
}

// ---------------- one circuit layer: sum or logsumexp over FAN=4 -----------
// Block-uniform int4 child index -> scalar s_load; gathers coalesce per row.
__global__ void circuit_layer_kernel(const float* __restrict__ prev,
                                     const int*  __restrict__ idx,
                                     float* __restrict__ cur, int isSum)
{
    const int n  = blockIdx.x;
    const int bb = threadIdx.x;
    const int4 c = ((const int4*)idx)[n];
    const float v0 = prev[(size_t)c.x * 256 + bb];
    const float v1 = prev[(size_t)c.y * 256 + bb];
    const float v2 = prev[(size_t)c.z * 256 + bb];
    const float v3 = prev[(size_t)c.w * 256 + bb];
    float r;
    if (isSum) {
        r = (v0 + v1) + (v2 + v3);
    } else {
        const float m = fmaxf(fmaxf(v0, v1), fmaxf(v2, v3));
        r = m + __logf(__expf(v0 - m) + __expf(v1 - m) +
                       __expf(v2 - m) + __expf(v3 - m));
    }
    cur[(size_t)n * 256 + bb] = r;
}

// ---------------- root (width 1, sum layer) --------------------------------
__global__ void root_kernel(const float* __restrict__ prev,
                            const int*  __restrict__ idx,
                            float* __restrict__ out)
{
    const int bb = threadIdx.x;
    const int4 c = *(const int4*)idx;
    out[bb] = prev[(size_t)c.x * 256 + bb] + prev[(size_t)c.y * 256 + bb]
            + prev[(size_t)c.z * 256 + bb] + prev[(size_t)c.w * 256 + bb];
}

// ---------------------------------------------------------------------------
extern "C" void kernel_launch(void* const* d_in, const int* in_sizes, int n_in,
                              void* d_out, int out_size, void* d_ws, size_t ws_size,
                              hipStream_t stream)
{
    (void)in_sizes; (void)n_in; (void)out_size; (void)ws_size;

    const float* images  = (const float*)d_in[0];
    const float* conv1_w = (const float*)d_in[1];
    const float* conv1_b = (const float*)d_in[2];
    const float* conv2_w = (const float*)d_in[3];
    const float* conv2_b = (const float*)d_in[4];
    const float* fc1_w   = (const float*)d_in[5];
    const float* fc1_b   = (const float*)d_in[6];
    const float* fc2_w   = (const float*)d_in[7];
    const float* fc2_b   = (const float*)d_in[8];
    const float* fc3_w   = (const float*)d_in[9];
    const float* fc3_b   = (const float*)d_in[10];
    const int* children[7];
    for (int i = 0; i < 7; ++i) children[i] = (const int*)d_in[11 + i];

    char* ws = (char*)d_ws;
    const size_t OFF_CBUF1 = 33554432;
    const size_t OFF_ACT2H = 71663616;               // end of ACT1 region
    const size_t OFF_FC1OH = OFF_ACT2H + 10616832;   // 20736*256*2
    const size_t OFF_FC2O  = OFF_FC1OH + 5308416;    // 20736*128*2
    const size_t OFF_WH1   = OFF_FC2O  + 6967296;    // 20736*84*4
    const size_t OFF_WH2   = OFF_WH1   + 65536;      // 128*256*2
    const size_t OFF_LP    = OFF_WH2   + 24576;      // 96*128*2

    float*     act1  = (float*)(ws + 0);             // aliases cbuf0/cbuf1
    _Float16*  act2h = (_Float16*)(ws + OFF_ACT2H);
    _Float16*  fc1oh = (_Float16*)(ws + OFF_FC1OH);
    float*     fc2o  = (float*)(ws + OFF_FC2O);
    _Float16*  wh1   = (_Float16*)(ws + OFF_WH1);
    _Float16*  wh2   = (_Float16*)(ws + OFF_WH2);
    float*     lp    = (float*)(ws + OFF_LP);
    float*     cbuf0 = (float*)(ws + 0);
    float*     cbuf1 = (float*)(ws + OFF_CBUF1);

    weights_to_f16_kernel<<<64, 256, 0, stream>>>(fc1_w, fc2_w, wh1, wh2);
    conv1_pool_kernel<<<NIMG, 256, 0, stream>>>(images, conv1_w, conv1_b, act1);
    conv2_pool_kernel<<<NIMG, 256, 0, stream>>>(act1, conv2_w, conv2_b, act2h);

    // FC1: K=256, N=120 -> padded out stride 128 (8 n-tiles, 8 waves/block)
    fc_wmma_kernel<256, 120, 128, true, _Float16>
        <<<NIMG / 16, 256, 0, stream>>>(act2h, wh1, fc1_b, fc1oh);
    // FC2: KPAD=128, N=84 -> f32 out stride 84 (6 n-tiles, 6 waves/block)
    fc_wmma_kernel<128, 84, 84, true, float>
        <<<NIMG / 16, 192, 0, stream>>>(fc1oh, wh2, fc2_b, fc2o);

    fc3_lsm_kernel<<<(NIMG + 255) / 256, 256, 0, stream>>>(fc2o, fc3_w, fc3_b, lp);

    leaf_kernel<<<1458, 256, 0, stream>>>(lp, cbuf0);

    const float* prev = cbuf0;
    float*       cur  = cbuf1;
    for (int i = 0; i < 6; ++i) {
        circuit_layer_kernel<<<32768, 256, 0, stream>>>(prev, children[i], cur,
                                                        (i % 2 == 0) ? 1 : 0);
        float* t = (float*)prev; prev = cur; cur = t;
    }
    root_kernel<<<1, 256, 0, stream>>>(prev, children[6], (float*)d_out);
}
